// Som_40063454937634
// MI455X (gfx1250) — compile-verified, matrix operand order checked by reading, not run on hardware
//
#include <hip/hip_runtime.h>
#include <stdint.h>

// ---------------------------------------------------------------------------
// SOM BMU search for MI455X (gfx1250, wave32, WMMA).
// score(b,m) = ||w_m||^2 - 2 * x_b . w_m   (||x_b||^2 dropped: row-constant)
// GEMM done with split-bf16 (hi+lo) 3-product emulation on
// v_wmma_f32_16x16x32_bf16 -> ~fp32 accuracy at bf16 matrix throughput.
// Per-row argmin via packed (orderable-f32-key | index) u64 global atomic min.
// ---------------------------------------------------------------------------

typedef __attribute__((ext_vector_type(16))) __bf16 v16bf;
typedef __attribute__((ext_vector_type(8)))  __bf16 v8bf;
typedef __attribute__((ext_vector_type(8)))  float  v8f;

#define SOM_ROWS  128
#define SOM_COLS  128
#define NFEAT     256
#define BATCH     4096
#define M_TOTAL   (SOM_ROWS * SOM_COLS)

#define BLK  128   // output tile (rows of xb) x (codebook entries)
#define BK   32    // K step (bf16 WMMA K)
#define LDT  40    // padded LDS row stride in bf16 elems (80B: 16B-aligned, bank-conflict-free)

// -------------------------------- helpers ----------------------------------

__device__ __forceinline__ void bsplit(float a, __bf16* h, __bf16* l) {
    __bf16 hh = (__bf16)a;              // RNE high part
    *h = hh;
    *l = (__bf16)(a - (float)hh);       // residual, ~2^-17 relative total
}

// Build a 16-element bf16 fragment from two 16-byte LDS chunks
// (elements 0..7 = first K-run, 8..15 = second K-run per ISA layout).
__device__ __forceinline__ v16bf ldfrag(const __bf16* p0, const __bf16* p1) {
    v8bf a = *(const v8bf*)p0;
    v8bf b = *(const v8bf*)p1;
    v16bf r;
#pragma unroll
    for (int i = 0; i < 8; ++i) { r[i] = a[i]; r[i + 8] = b[i]; }
    return r;
}

__device__ __forceinline__ unsigned long long pack_key(float s, unsigned idx) {
    unsigned u = __float_as_uint(s);
    u = (u & 0x80000000u) ? ~u : (u | 0x80000000u);   // monotone float->uint
    return ((unsigned long long)u << 32) | idx;        // ties -> smaller index
}

// ------------------------------ kernels ------------------------------------

__global__ void som_init(unsigned long long* __restrict__ mins) {
    int i = blockIdx.x * blockDim.x + threadIdx.x;
    if (i < BATCH) mins[i] = 0xFFFFFFFFFFFFFFFFULL;
}

// one wave per codebook entry: exact f32 ||w_m||^2
__global__ void som_b2(const float* __restrict__ w, float* __restrict__ b2) {
    int wave = (blockIdx.x * blockDim.x + threadIdx.x) >> 5;
    int lane = threadIdx.x & 31;
    const float4* row = (const float4*)(w + (size_t)wave * NFEAT);
    float s = 0.f;
#pragma unroll
    for (int i = 0; i < 2; ++i) {
        float4 v = row[lane * 2 + i];
        s += v.x * v.x + v.y * v.y + v.z * v.z + v.w * v.w;
    }
#pragma unroll
    for (int off = 16; off; off >>= 1) s += __shfl_xor(s, off, 32);
    if (lane == 0) b2[wave] = s;
}

__global__ __launch_bounds__(256)
void som_gemm(const float* __restrict__ xb, const float* __restrict__ w,
              const float* __restrict__ b2,
              unsigned long long* __restrict__ mins) {
    __shared__ alignas(16) __bf16 Ah[BLK][LDT];
    __shared__ alignas(16) __bf16 Al[BLK][LDT];
    __shared__ alignas(16) __bf16 Bh[BLK][LDT];
    __shared__ alignas(16) __bf16 Bl[BLK][LDT];

    const int tid  = threadIdx.x;
    const int lane = tid & 31;
    const int wv   = tid >> 5;          // 8 waves
    const int R0   = (wv >> 1) * 32;    // wave's row base within tile
    const int C0   = (wv & 1) * 64;     // wave's col base within tile
    const int l15  = lane & 15;
    const int koff = (lane >> 4) * 8;   // per-lane K chunk select (ISA layout)

    const int gRow0 = blockIdx.y * BLK; // batch rows
    const int gCol0 = blockIdx.x * BLK; // codebook entries

    v8f acc[2][4];
#pragma unroll
    for (int rt = 0; rt < 2; ++rt)
#pragma unroll
        for (int ct = 0; ct < 4; ++ct) acc[rt][ct] = (v8f)(0.f);

    for (int k0 = 0; k0 < NFEAT; k0 += BK) {
        __syncthreads();
        // Stage tiles: f32 global -> split bf16 hi/lo in LDS.
        // A holds -2*x so score = acc + ||w||^2.
#pragma unroll
        for (int i = 0; i < 4; ++i) {
            int lin = i * 256 + tid;        // 1024 float4 slots per operand
            int r = lin >> 3;
            int c = (lin & 7) << 2;
            float4 va = *(const float4*)(xb + (size_t)(gRow0 + r) * NFEAT + k0 + c);
            float4 vb = *(const float4*)(w  + (size_t)(gCol0 + r) * NFEAT + k0 + c);
            bsplit(-2.0f * va.x, &Ah[r][c + 0], &Al[r][c + 0]);
            bsplit(-2.0f * va.y, &Ah[r][c + 1], &Al[r][c + 1]);
            bsplit(-2.0f * va.z, &Ah[r][c + 2], &Al[r][c + 2]);
            bsplit(-2.0f * va.w, &Ah[r][c + 3], &Al[r][c + 3]);
            bsplit(vb.x, &Bh[r][c + 0], &Bl[r][c + 0]);
            bsplit(vb.y, &Bh[r][c + 1], &Bl[r][c + 1]);
            bsplit(vb.z, &Bh[r][c + 2], &Bl[r][c + 2]);
            bsplit(vb.w, &Bh[r][c + 3], &Bl[r][c + 3]);
        }
        __syncthreads();

        // Fragment loads (two 16B LDS reads each; padded stride -> no bank conflicts)
        v16bf afh[2], afl[2];
#pragma unroll
        for (int rt = 0; rt < 2; ++rt) {
            const __bf16* ph = &Ah[R0 + rt * 16 + l15][koff];
            const __bf16* pl = &Al[R0 + rt * 16 + l15][koff];
            afh[rt] = ldfrag(ph, ph + 16);
            afl[rt] = ldfrag(pl, pl + 16);
        }
        v16bf bfh[4], bfl[4];
#pragma unroll
        for (int ct = 0; ct < 4; ++ct) {
            const __bf16* ph = &Bh[C0 + ct * 16 + l15][koff];
            const __bf16* pl = &Bl[C0 + ct * 16 + l15][koff];
            bfh[ct] = ldfrag(ph, ph + 16);
            bfl[ct] = ldfrag(pl, pl + 16);
        }

        // Split-bf16 GEMM: hi*hi + hi*lo + lo*hi (lo*lo ~2^-18, dropped)
#pragma unroll
        for (int rt = 0; rt < 2; ++rt)
#pragma unroll
            for (int ct = 0; ct < 4; ++ct) {
                v8f c = acc[rt][ct];
                c = __builtin_amdgcn_wmma_f32_16x16x32_bf16(
                        false, afh[rt], false, bfh[ct], (short)0, c, false, false);
                c = __builtin_amdgcn_wmma_f32_16x16x32_bf16(
                        false, afh[rt], false, bfl[ct], (short)0, c, false, false);
                c = __builtin_amdgcn_wmma_f32_16x16x32_bf16(
                        false, afl[rt], false, bfh[ct], (short)0, c, false, false);
                acc[rt][ct] = c;
            }
    }

    // Epilogue: add ||w||^2, pack (key,index), reduce over 64 cols, atomic-min.
#pragma unroll
    for (int rt = 0; rt < 2; ++rt) {
        unsigned long long best[8];
#pragma unroll
        for (int r = 0; r < 8; ++r) best[r] = 0xFFFFFFFFFFFFFFFFULL;
#pragma unroll
        for (int ct = 0; ct < 4; ++ct) {
            int col = gCol0 + C0 + ct * 16 + l15;     // global codebook index
            float bb = b2[col];
#pragma unroll
            for (int r = 0; r < 8; ++r) {
                unsigned long long p = pack_key(acc[rt][ct][r] + bb, (unsigned)col);
                if (p < best[r]) best[r] = p;
            }
        }
#pragma unroll
        for (int r = 0; r < 8; ++r) {
            unsigned long long b = best[r];
            // xor 1,2,4,8 stays within the 16-lane half that owns this row set
#pragma unroll
            for (int off = 1; off <= 8; off <<= 1) {
                unsigned hi = (unsigned)(b >> 32), lo = (unsigned)b;
                unsigned ohi = __shfl_xor(hi, off, 32);
                unsigned olo = __shfl_xor(lo, off, 32);
                unsigned long long o = ((unsigned long long)ohi << 32) | olo;
                if (o < b) b = o;
            }
            if (l15 == 0) {
                int row = gRow0 + R0 + rt * 16 + r + ((lane >> 4) ? 8 : 0);
                atomicMin(&mins[row], b);
            }
        }
    }
}

__global__ void som_final(const unsigned long long* __restrict__ mins,
                          int* __restrict__ out) {
    int i = blockIdx.x * blockDim.x + threadIdx.x;
    if (i < BATCH) {
        unsigned idx = (unsigned)(mins[i] & 0xFFFFFFFFu);
        out[2 * i + 0] = (int)(idx >> 7);     // / COLS (128)
        out[2 * i + 1] = (int)(idx & 127u);   // % COLS
    }
}

// ------------------------------ launcher -----------------------------------

extern "C" void kernel_launch(void* const* d_in, const int* in_sizes, int n_in,
                              void* d_out, int out_size, void* d_ws, size_t ws_size,
                              hipStream_t stream) {
    (void)in_sizes; (void)n_in; (void)out_size; (void)ws_size;
    const float* xb = (const float*)d_in[0];   // (4096, 256) f32
    const float* w  = (const float*)d_in[1];   // (128,128,256) f32 == (16384,256)

    unsigned long long* mins = (unsigned long long*)d_ws;              // 32 KB
    float* b2 = (float*)((char*)d_ws + (size_t)BATCH * sizeof(unsigned long long)); // 64 KB

    som_init<<<BATCH / 256, 256, 0, stream>>>(mins);
    som_b2<<<M_TOTAL / 8, 256, 0, stream>>>(w, b2);   // 8 waves/block, 1 entry/wave

    dim3 grid(M_TOTAL / BLK, BATCH / BLK);            // (128, 32)
    som_gemm<<<grid, 256, 0, stream>>>(xb, w, b2, mins);

    som_final<<<BATCH / 256, 256, 0, stream>>>(mins, (int*)d_out);
}
